// CausalSelfAttention_20718922236626
// MI455X (gfx1250) — matrix operand order repaired and checked by reading, not compiled, not run on hardware
//
#include <hip/hip_runtime.h>

#define BB 2
#define NN 4096
#define DD 768
#define HH 12
#define HD 64
#define E3 192  /* 3*HD */

typedef _Float16 f16_t;
typedef __attribute__((ext_vector_type(16))) _Float16 v16h;
typedef __attribute__((ext_vector_type(8)))  _Float16 v8h;
typedef __attribute__((ext_vector_type(8)))  float    v8f;

union V16U { v16h v; v8h h[2]; };

static __device__ inline v8f wmma_f16(v16h a, v16h b, v8f c) {
  // D = A(16x32 f16) * B(32x16 f16) + C(16x16 f32)
  return __builtin_amdgcn_wmma_f32_16x16x32_f16(false, a, false, b, (short)0, c, false, false);
}

// A fragment: row-major matrix, rows row0..row0+15, K columns kb..kb+31.
// ISA layout: lane m = lane&15; lanes<16 hold K {kb+0..7, kb+16..23},
// lanes>=16 hold K {kb+8..15, kb+24..31}.
static __device__ inline v16h load_a_frag(const f16_t* base, int lda, int row0, int kb, int lane) {
  int row  = row0 + (lane & 15);
  int koff = (lane >> 4) << 3;
  const f16_t* p = base + (long)row * lda + kb + koff;
  V16U u;
  u.h[0] = *(const v8h*)(p);
  u.h[1] = *(const v8h*)(p + 16);
  return u.v;
}

// B fragment: "column-major" storage base[col*ldb + k] (K contiguous per column).
// ISA layout: col n = lane&15; lanes<16 hold K {kb+0..7, kb+16..23}, lanes>=16 hold K {kb+8..15, kb+24..31}.
static __device__ inline v16h load_b_frag(const f16_t* base, int ldb, int col0, int kb, int lane) {
  int col  = col0 + (lane & 15);
  int koff = (lane >> 4) << 3;
  const f16_t* p = base + (long)col * ldb + kb + koff;
  V16U u;
  u.h[0] = *(const v8h*)(p);
  u.h[1] = *(const v8h*)(p + 16);
  return u.v;
}

/* ---------------- prep: f32 -> f16 conversions ---------------- */

__global__ void k_cvt(const float* __restrict__ src, f16_t* __restrict__ dst, int n) {
  int i = blockIdx.x * blockDim.x + threadIdx.x;
  if (i < n) dst[i] = (f16_t)src[i];
}

// W_kqv [H, D, 192] f32 -> wkt [H, 192, D] f16 (transpose so K=D is contiguous per out-col)
__global__ void k_cvt_wkqv(const float* __restrict__ w, f16_t* __restrict__ wkt) {
  int i = blockIdx.x * blockDim.x + threadIdx.x;
  if (i < HH * DD * E3) {
    int h = i / (DD * E3);
    int r = i % (DD * E3);
    int d = r / E3;
    int e = r % E3;
    wkt[((long)h * E3 + e) * DD + d] = (f16_t)w[i];
  }
}

/* ---------------- KQV projection ----------------
   per wave: rows n0..n0+15, one 64-wide group g (0=k,1=q,2=v) */
__global__ __launch_bounds__(128) void k_kqv(const f16_t* __restrict__ xb,
                                             const f16_t* __restrict__ wkt,
                                             f16_t* __restrict__ qbuf,
                                             f16_t* __restrict__ kbuf,
                                             f16_t* __restrict__ vT) {
  int wid  = (blockIdx.x * blockDim.x + threadIdx.x) >> 5;
  int lane = threadIdx.x & 31;
  int g  = wid % 3;
  int nt = (wid / 3) % (NN / 16);
  int bh = wid / (3 * (NN / 16));
  if (bh >= BB * HH) return;
  int b = bh / HH, h = bh % HH;

  const f16_t* A  = xb  + (long)b * NN * DD;
  const f16_t* Bm = wkt + (long)h * E3 * DD;
  int n0 = nt * 16;
  int e0 = g * 64;

  v8f c0 = {}, c1 = {}, c2 = {}, c3 = {};
  for (int kk = 0; kk < DD; kk += 32) {
    v16h a = load_a_frag(A, DD, n0, kk, lane);
    c0 = wmma_f16(a, load_b_frag(Bm, DD, e0 +  0, kk, lane), c0);
    c1 = wmma_f16(a, load_b_frag(Bm, DD, e0 + 16, kk, lane), c1);
    c2 = wmma_f16(a, load_b_frag(Bm, DD, e0 + 32, kk, lane), c2);
    c3 = wmma_f16(a, load_b_frag(Bm, DD, e0 + 48, kk, lane), c3);
  }

  int colq  = lane & 15;
  int half8 = (lane >> 4) << 3;
  v8f cc[4] = {c0, c1, c2, c3};
  for (int cf = 0; cf < 4; ++cf) {
    int d = cf * 16 + colq;  // 0..63 within group
    for (int r = 0; r < 8; ++r) {
      int row = n0 + half8 + r;
      f16_t v = (f16_t)cc[cf][r];
      if (g == 0)      kbuf[((long)bh * NN + row) * HD + d] = v;
      else if (g == 1) qbuf[((long)bh * NN + row) * HD + d] = v;
      else             vT[((long)bh * HD + d) * NN + row] = v;   // transposed
    }
  }
}

/* ---------------- flash attention (transposed tiles) ----------------
   per wave: 16 queries; S^T = K*Q^T so softmax stats are per lane. */
__global__ __launch_bounds__(128) void k_attn(const f16_t* __restrict__ qbuf,
                                              const f16_t* __restrict__ kbuf,
                                              const f16_t* __restrict__ vT,
                                              f16_t* __restrict__ ao) {
  __shared__ __align__(32) f16_t smem[4 * 16 * 32];
  int winb = threadIdx.x >> 5;
  int wid  = blockIdx.x * 4 + winb;
  int lane = threadIdx.x & 31;
  int qt = wid % (NN / 16);
  int bh = wid / (NN / 16);
  if (bh >= BB * HH) return;
  int q0 = qt * 16;

  const f16_t* qh = qbuf + (long)bh * NN * HD;
  const f16_t* kh = kbuf + (long)bh * NN * HD;
  const f16_t* vh = vT   + (long)bh * HD * NN;
  f16_t* sm = smem + winb * (16 * 32);

  // loop-invariant Q^T B-fragments (K = d in [0,32) and [32,64))
  v16h qf0 = load_b_frag(qh + (long)q0 * HD, HD, 0, 0,  lane);
  v16h qf1 = load_b_frag(qh + (long)q0 * HD, HD, 0, 32, lane);

  const float MASK = -1e30f;
  float mrow = MASK, lrow = 0.f;
  v8f o0 = {}, o1 = {}, o2 = {}, o3 = {};
  int colq  = lane & 15;
  int half8 = (lane >> 4) << 3;
  int kend = q0 + 16;  // keys 0..q0+15

  for (int kc = 0; kc < kend; kc += 32) {
    v8f s0 = {}, s1 = {};
    {  // sub-block 0: keys kc..kc+15  (kc <= q0, always in-bounds)
      v16h a0 = load_a_frag(kh, HD, kc, 0, lane);
      v16h a1 = load_a_frag(kh, HD, kc, 32, lane);
      v8f c = {};
      c = wmma_f16(a0, qf0, c);
      c = wmma_f16(a1, qf1, c);
      for (int r = 0; r < 8; ++r) {
        float sv = c[r] * 0.125f;                 // 1/sqrt(64)
        int key = kc + half8 + r;
        if (key > q0 + colq) sv = MASK;           // causal (only fires near diagonal)
        s0[r] = sv;
      }
    }
    int kk1 = kc + 16;
    if (kk1 < kend) {  // sub-block 1
      v16h a0 = load_a_frag(kh, HD, kk1, 0, lane);
      v16h a1 = load_a_frag(kh, HD, kk1, 32, lane);
      v8f c = {};
      c = wmma_f16(a0, qf0, c);
      c = wmma_f16(a1, qf1, c);
      for (int r = 0; r < 8; ++r) {
        float sv = c[r] * 0.125f;
        int key = kk1 + half8 + r;
        if (key > q0 + colq) sv = MASK;
        s1[r] = sv;
      }
    } else {
      for (int r = 0; r < 8; ++r) s1[r] = MASK;
    }

    // online softmax (per lane = per query; combine the two lane-halves)
    float mloc = MASK;
    for (int r = 0; r < 8; ++r) { mloc = fmaxf(mloc, s0[r]); mloc = fmaxf(mloc, s1[r]); }
    mloc = fmaxf(mloc, __shfl_xor(mloc, 16, 32));
    float mnew  = fmaxf(mrow, mloc);
    float alpha = __expf(mrow - mnew);
    float rs = 0.f;
    v8h p0, p1;
    for (int r = 0; r < 8; ++r) {
      float e0v = __expf(s0[r] - mnew);
      float e1v = __expf(s1[r] - mnew);
      rs += e0v + e1v;
      p0[r] = (f16_t)e0v;
      p1[r] = (f16_t)e1v;
    }
    rs += __shfl_xor(rs, 16, 32);
    lrow = lrow * alpha + rs;
    mrow = mnew;
    for (int r = 0; r < 8; ++r) { o0[r] *= alpha; o1[r] *= alpha; o2[r] *= alpha; o3[r] *= alpha; }

    // relayout P (C-frag) -> P^T B-frag via per-wave LDS scratch [q][key 0..31]
    *(v8h*)(sm + colq * 32 + half8)      = p0;
    *(v8h*)(sm + colq * 32 + 16 + half8) = p1;
    asm volatile("" ::: "memory");
    V16U pu;
    pu.h[0] = *(const v8h*)(sm + colq * 32 + half8);
    pu.h[1] = *(const v8h*)(sm + colq * 32 + 16 + half8);
    v16h pf = pu.v;
    asm volatile("" ::: "memory");

    // O^T += V^T(64xK) * P^T(Kx16), K = 32 keys
    o0 = wmma_f16(load_a_frag(vh, NN,  0, kc, lane), pf, o0);
    o1 = wmma_f16(load_a_frag(vh, NN, 16, kc, lane), pf, o1);
    o2 = wmma_f16(load_a_frag(vh, NN, 32, kc, lane), pf, o2);
    o3 = wmma_f16(load_a_frag(vh, NN, 48, kc, lane), pf, o3);
  }

  // write merged-head output: ao[b][q0+q][h*64 + d]; lane holds query colq, rows d
  int b = bh / HH, h = bh % HH;
  float inv = 1.0f / lrow;
  f16_t* orow = ao + ((long)b * NN + q0 + colq) * DD + h * HD;
  v8f oo[4] = {o0, o1, o2, o3};
  for (int dblk = 0; dblk < 4; ++dblk)
    for (int r = 0; r < 8; ++r)
      orow[dblk * 16 + half8 + r] = (f16_t)(oo[dblk][r] * inv);
}

/* ---------------- output projection: out = ao @ Wproj^T + b ---------------- */
__global__ __launch_bounds__(128) void k_proj(const f16_t* __restrict__ ao,
                                              const f16_t* __restrict__ wp,
                                              const float* __restrict__ bias,
                                              float* __restrict__ out) {
  int wid  = (blockIdx.x * blockDim.x + threadIdx.x) >> 5;
  int lane = threadIdx.x & 31;
  int jb = wid % (DD / 64);
  int tt = wid / (DD / 64);
  if (tt >= (BB * NN) / 16) return;
  int t0 = tt * 16, j0 = jb * 64;

  v8f c0 = {}, c1 = {}, c2 = {}, c3 = {};
  for (int kk = 0; kk < DD; kk += 32) {
    v16h a = load_a_frag(ao, DD, t0, kk, lane);
    c0 = wmma_f16(a, load_b_frag(wp, DD, j0 +  0, kk, lane), c0);
    c1 = wmma_f16(a, load_b_frag(wp, DD, j0 + 16, kk, lane), c1);
    c2 = wmma_f16(a, load_b_frag(wp, DD, j0 + 32, kk, lane), c2);
    c3 = wmma_f16(a, load_b_frag(wp, DD, j0 + 48, kk, lane), c3);
  }
  int colq = lane & 15, half8 = (lane >> 4) << 3;
  v8f cc[4] = {c0, c1, c2, c3};
  for (int cf = 0; cf < 4; ++cf) {
    int j = j0 + cf * 16 + colq;
    float bj = bias[j];
    for (int r = 0; r < 8; ++r)
      out[((long)t0 + half8 + r) * DD + j] = cc[cf][r] + bj;
  }
}

extern "C" void kernel_launch(void* const* d_in, const int* in_sizes, int n_in,
                              void* d_out, int out_size, void* d_ws, size_t ws_size,
                              hipStream_t stream) {
  const float* x      = (const float*)d_in[0];  // [B,N,D]
  const float* W_kqv  = (const float*)d_in[1];  // [H,D,192]
  const float* W_proj = (const float*)d_in[2];  // [D,D]
  const float* b_proj = (const float*)d_in[3];  // [D]
  float* out = (float*)d_out;

  const long n_x  = (long)BB * NN * DD;      // 6291456
  const long n_wk = (long)HH * DD * E3;      // 1769472
  const long n_wp = (long)DD * DD;           // 589824
  const long n_hd = (long)BB * HH * NN * HD; // 6291456

  f16_t* xb   = (f16_t*)d_ws;
  f16_t* wkt  = xb   + n_x;
  f16_t* wp   = wkt  + n_wk;
  f16_t* qbuf = wp   + n_wp;
  f16_t* kbuf = qbuf + n_hd;
  f16_t* vT   = kbuf + n_hd;
  f16_t* ao   = vT   + n_hd;   // total ~65 MB of workspace

  k_cvt<<<(int)((n_x + 255) / 256), 256, 0, stream>>>(x, xb, (int)n_x);
  k_cvt_wkqv<<<(int)((n_wk + 255) / 256), 256, 0, stream>>>(W_kqv, wkt);
  k_cvt<<<(int)((n_wp + 255) / 256), 256, 0, stream>>>(W_proj, wp, (int)n_wp);

  {  // KQV: waves = B*H*(N/16)*3 = 18432, 4 waves/block
    int waves = BB * HH * (NN / 16) * 3;
    k_kqv<<<waves / 4, 128, 0, stream>>>(xb, wkt, qbuf, kbuf, vT);
  }
  {  // attention: waves = B*H*(N/16) = 6144
    int waves = BB * HH * (NN / 16);
    k_attn<<<waves / 4, 128, 0, stream>>>(qbuf, kbuf, vT, ao);
  }
  {  // projection: waves = (B*N/16)*(D/64) = 6144
    int waves = ((BB * NN) / 16) * (DD / 64);
    k_proj<<<waves / 4, 128, 0, stream>>>(ao, wp, b_proj, out);
  }
}
